// RMVSNet_7937099563153
// MI455X (gfx1250) — matrix operand order repaired and checked by reading, not compiled, not run on hardware
//
#include <hip/hip_runtime.h>
#include <hip/hip_bf16.h>
#include <math.h>

// ---------------------------------------------------------------------------
// MI455X (gfx1250) implementation of RMVSNet forward.
//  * All convolutions (UNet + GRU + prob) run as implicit-GEMM on
//    v_wmma_f32_16x16x32_bf16 (wave32, 16x16 C/D tiles, K chunks of 32).
//  * Per-block weight slab (16 cout rows x KKpad) is DMA'd into LDS once by
//    the Tensor Data Mover (tensor_load_to_lds + s_wait_tensorcnt), with
//    hardware OOB zero-fill providing the K / Cout padding.
//  * A-operand im2col gathers are branchless (clamped address + cndmask).
// ---------------------------------------------------------------------------

typedef __attribute__((ext_vector_type(16))) __bf16        v16bf;
typedef __attribute__((ext_vector_type(8)))  float         v8f;
typedef __attribute__((ext_vector_type(4)))  unsigned int  v4u;
typedef __attribute__((ext_vector_type(8)))  int           v8i;
typedef __attribute__((ext_vector_type(4)))  int           v4i;

static inline int cdiv_host(long long a, long long b) { return (int)((a + b - 1) / b); }

// ---------------------------------------------------------------------------
// f32 -> bf16 convert / zero fills
// ---------------------------------------------------------------------------
__global__ void cvt_f32_bf16(const float* __restrict__ x, __bf16* __restrict__ y, size_t n) {
  size_t i = (size_t)blockIdx.x * blockDim.x + threadIdx.x;
  if (i < n) y[i] = (__bf16)x[i];
}

__global__ void zero_f32(float* __restrict__ x, size_t n) {
  size_t i = (size_t)blockIdx.x * blockDim.x + threadIdx.x;
  if (i < n) x[i] = 0.0f;
}

__global__ void zero_bf16(__bf16* __restrict__ x, size_t n) {
  size_t i = (size_t)blockIdx.x * blockDim.x + threadIdx.x;
  if (i < n) x[i] = (__bf16)0.0f;
}

// ---------------------------------------------------------------------------
// Implicit-GEMM conv via WMMA + TDM weight staging.
//   in  : (N, Cin, H, W) bf16
//   wgt : (Cout, Cin, KSZ, KSZ) bf16  (flat K index = cin*KSZ*KSZ + ky*KSZ + kx)
//   out : (N, Cout, Ho, Wo) f32  (+bias if non-null)
// MODE 0: normal conv, stride s, 'pad' = begin padding (JAX-SAME).
// MODE 1: conv_transpose stride 2 (fractionally-strided gather), pad = 1.
// Grid: (Ho*Wo/64, ceil(Cout/16), N), block 128 (4 waves). Ho*Wo % 64 == 0.
// ---------------------------------------------------------------------------
template <int KSZ, int MODE>
__global__ __launch_bounds__(128)
void conv2d_wmma(const __bf16* __restrict__ in, const __bf16* __restrict__ wgt,
                 const float* __restrict__ bias, float* __restrict__ out,
                 int Cin, int H, int W, int Cout, int stride, int pad,
                 int Ho, int Wo) {
  constexpr int KSQ = KSZ * KSZ;
  // Weight slab: 16 cout rows x KKpad (KK up to 128*9=1152, padded to 32).
  __shared__ __align__(16) __bf16 Bs[16 * 1152];

  const int lane = threadIdx.x & 31;
  const int wave = threadIdx.x >> 5;
  const int hi   = lane >> 4;   // 0: lanes 0-15, 1: lanes 16-31
  const int lo   = lane & 15;
  const int HoWo = Ho * Wo;
  const int mBase    = (blockIdx.x * 4 + wave) * 16;
  const int coutBase = blockIdx.y * 16;
  const int n        = blockIdx.z;
  const int KK    = Cin * KSQ;
  const int KKpad = (KK + 31) & ~31;
  const size_t inBase = (size_t)n * Cin * H * W;

  // ---- TDM: DMA the (16 x KKpad) weight tile into LDS (wave 0 issues). ----
  if (wave == 0) {
    unsigned long long ga = (unsigned long long)(uintptr_t)(wgt + (size_t)coutBase * KK);
    unsigned int lds = (unsigned int)(uintptr_t)(&Bs[0]);
    int rows = Cout - coutBase;  // tensor_dim1: rows past this read back zero
    if (rows > 16) rows = 16;

    v4u g0;
    g0.x = 1u;                                   // count=1 user descriptor
    g0.y = lds;                                  // lds_addr (bytes)
    g0.z = (unsigned int)ga;                     // global_addr[31:0]
    g0.w = (unsigned int)((ga >> 32) & 0x01FFFFFFu) | (2u << 30);  // [56:32] | type=2

    // Group1 bitfields (256 bits, little-endian):
    //  [17:16]=data_size(1 -> 2B)  [79:48]=tensor_dim0=KK  [111:80]=tensor_dim1=rows
    //  [127:112]=tile_dim0=KKpad   [143:128]=tile_dim1=16  [207:160]=dim0_stride=KK
    unsigned long long q0 = (1ull << 16) | ((unsigned long long)(KK & 0xFFFF) << 48);
    unsigned long long q1 = (unsigned long long)((KK >> 16) & 0xFFFF)
                          | ((unsigned long long)(unsigned int)rows << 16)
                          | ((unsigned long long)(KKpad & 0xFFFF) << 48);
    unsigned long long q2 = 16ull | ((unsigned long long)(unsigned int)KK << 32);
    unsigned long long q3 = 0ull;
    v8i g1;
    g1[0] = (int)(unsigned int)q0; g1[1] = (int)(unsigned int)(q0 >> 32);
    g1[2] = (int)(unsigned int)q1; g1[3] = (int)(unsigned int)(q1 >> 32);
    g1[4] = (int)(unsigned int)q2; g1[5] = (int)(unsigned int)(q2 >> 32);
    g1[6] = (int)(unsigned int)q3; g1[7] = (int)(unsigned int)(q3 >> 32);
    v4i g2 = {0, 0, 0, 0};
    v4i g3 = {0, 0, 0, 0};
#if defined(__clang_major__) && (__clang_major__ >= 23)
    v8i g4 = {0, 0, 0, 0, 0, 0, 0, 0};
    __builtin_amdgcn_tensor_load_to_lds(g0, g1, g2, g3, g4, 0);
#else
    __builtin_amdgcn_tensor_load_to_lds(g0, g1, g2, g3, 0);
#endif
    __builtin_amdgcn_s_wait_tensorcnt(0);
  }
  __syncthreads();

  // Per-lane output pixel coordinates (A-matrix row M = lo).
  const int m  = mBase + lo;
  const int oy = m / Wo;
  const int ox = m - oy * Wo;

  v8f acc = {0.f, 0.f, 0.f, 0.f, 0.f, 0.f, 0.f, 0.f};

  for (int kBase = 0; kBase < KK; kBase += 32) {
    // A fragment (branchless im2col gather): lane L holds M = L%16;
    // K offsets 0..15 (lanes<16) or 16..31 (lanes>=16).
    v16bf a;
#pragma unroll
    for (int i = 0; i < 16; ++i) {
      int kIdx = kBase + i + 16 * hi;
      int kcl  = (kIdx < KK - 1) ? kIdx : (KK - 1);
      int c  = kcl / KSQ;
      int r  = kcl - c * KSQ;
      int ky = r / KSZ;
      int kx = r - ky * KSZ;
      int iy, ix;
      bool ok = kIdx < KK;
      if (MODE == 0) {
        iy = oy * stride + ky - pad;
        ix = ox * stride + kx - pad;
      } else {
        int ty = oy + pad - ky;
        int tx = ox + pad - kx;
        ok = ok & ((ty & 1) == 0) & ((tx & 1) == 0);
        iy = ty >> 1;
        ix = tx >> 1;
      }
      ok = ok & (iy >= 0) & (iy < H) & (ix >= 0) & (ix < W);
      int iyc = iy < 0 ? 0 : (iy >= H ? H - 1 : iy);
      int ixc = ix < 0 ? 0 : (ix >= W ? W - 1 : ix);
      __bf16 v = in[inBase + ((size_t)c * H + iyc) * W + ixc];
      a[i] = ok ? v : (__bf16)0.0f;
    }

    // B fragment: lane L holds N = L%16, K = i + 16*(L>=16).
    // LDS slab is [row n][k] with row stride KKpad -> two 16B ds loads.
    union { v16bf v; uint4 q[2]; } bu;
    const uint4* bp = (const uint4*)(Bs + (size_t)lo * KKpad + kBase + 16 * hi);
    bu.q[0] = bp[0];
    bu.q[1] = bp[1];

    acc = __builtin_amdgcn_wmma_f32_16x16x32_bf16(
        /*neg_a=*/false, a, /*neg_b=*/false, bu.v,
        /*c_mod=*/(short)0, acc, /*reuse_a=*/false, /*reuse_b=*/false);
  }

  // Store: C/D layout -> lane holds channel (coutBase+lo), rows m = mBase+8*hi+r.
  const int co = coutBase + lo;
  if (co < Cout) {
    float bv = bias ? bias[co] : 0.0f;
    float* dst = out + ((size_t)(n * Cout + co)) * HoWo + mBase + 8 * hi;
    float4 w0 = make_float4(acc[0] + bv, acc[1] + bv, acc[2] + bv, acc[3] + bv);
    float4 w1 = make_float4(acc[4] + bv, acc[5] + bv, acc[6] + bv, acc[7] + bv);
    *(float4*)(dst)     = w0;
    *(float4*)(dst + 4) = w1;
  }
}

// ---------------------------------------------------------------------------
// GroupNorm statistics: one block per (n, group); group size is C/G channels.
// ---------------------------------------------------------------------------
__global__ __launch_bounds__(256)
void gn_stats_kernel(const float* __restrict__ x, float* __restrict__ mu,
                     float* __restrict__ rstd, int C, int HW, int G) {
  __shared__ float ss[256];
  __shared__ float sq[256];
  const int bid = blockIdx.x;
  const int n   = bid / G;
  const int g   = bid % G;
  const int cg  = C / G;
  const float* base = x + ((size_t)n * C + (size_t)g * cg) * HW;
  const size_t cnt  = (size_t)cg * HW;
  float s = 0.f, q = 0.f;
  for (size_t i = threadIdx.x; i < cnt; i += 256) {
    float v = base[i];
    s += v; q += v * v;
  }
  ss[threadIdx.x] = s;
  sq[threadIdx.x] = q;
  __syncthreads();
  for (int st = 128; st > 0; st >>= 1) {
    if ((int)threadIdx.x < st) {
      ss[threadIdx.x] += ss[threadIdx.x + st];
      sq[threadIdx.x] += sq[threadIdx.x + st];
    }
    __syncthreads();
  }
  if (threadIdx.x == 0) {
    float m = ss[0] / (float)cnt;
    float v = sq[0] / (float)cnt - m * m;
    mu[bid]   = m;
    rstd[bid] = rsqrtf(v + 1e-5f);
  }
}

// Fused GN-apply + ReLU + bf16 convert (bf16 is what the next conv consumes).
__global__ void gn_apply_relu(const float* __restrict__ x, const float* __restrict__ gamma,
                              const float* __restrict__ beta, const float* __restrict__ mu,
                              const float* __restrict__ rstd, __bf16* __restrict__ y,
                              int C, int HW, int G, size_t tot) {
  size_t i = (size_t)blockIdx.x * blockDim.x + threadIdx.x;
  if (i >= tot) return;
  int c = (int)((i / HW) % C);
  int n = (int)(i / ((size_t)C * HW));
  int cg = C / G;
  int st = n * G + c / cg;
  float v = (x[i] - mu[st]) * rstd[st] * gamma[c] + beta[c];
  y[i] = (__bf16)fmaxf(v, 0.0f);
}

// ---------------------------------------------------------------------------
// Channel concat (bf16, NCHW): out[n, 0:C1] = a, out[n, C1:C1+C2] = b.
// ---------------------------------------------------------------------------
__global__ void concat2_bf16(const __bf16* __restrict__ a, const __bf16* __restrict__ b,
                             __bf16* __restrict__ o, int C1, int C2, int HW, size_t tot) {
  size_t i = (size_t)blockIdx.x * blockDim.x + threadIdx.x;
  if (i >= tot) return;
  int C = C1 + C2;
  int hw = (int)(i % HW);
  int c  = (int)((i / HW) % C);
  int n  = (int)(i / ((size_t)C * HW));
  __bf16 v = (c < C1) ? a[((size_t)n * C1 + c) * HW + hw]
                      : b[((size_t)n * C2 + (c - C1)) * HW + hw];
  o[i] = v;
}

// ---------------------------------------------------------------------------
// Homographies: one thread per (view n, depth d). Hs layout: (3, D, 3, 3).
// ---------------------------------------------------------------------------
__device__ inline void mat3mul(const float* A, const float* B, float* C) {
#pragma unroll
  for (int i = 0; i < 3; ++i)
#pragma unroll
    for (int j = 0; j < 3; ++j)
      C[i * 3 + j] = A[i * 3 + 0] * B[0 * 3 + j] + A[i * 3 + 1] * B[1 * 3 + j] +
                     A[i * 3 + 2] * B[2 * 3 + j];
}

__device__ inline void mat3inv(const float* M, float* R) {
  float a = M[0], b = M[1], c = M[2];
  float d = M[3], e = M[4], f = M[5];
  float g = M[6], h = M[7], i = M[8];
  float A =  (e * i - f * h), B = -(d * i - f * g), C =  (d * h - e * g);
  float det = a * A + b * B + c * C;
  float inv = 1.0f / det;
  R[0] = A * inv;               R[1] = -(b * i - c * h) * inv; R[2] =  (b * f - c * e) * inv;
  R[3] = B * inv;               R[4] =  (a * i - c * g) * inv; R[5] = -(a * f - c * d) * inv;
  R[6] = C * inv;               R[7] = -(a * h - b * g) * inv; R[8] =  (a * e - b * d) * inv;
}

__global__ void homog_kernel(const float* __restrict__ Km, const float* __restrict__ E,
                             const float* __restrict__ ds, const float* __restrict__ di,
                             float* __restrict__ Hs, int NV, int D) {
  int t = blockIdx.x * blockDim.x + threadIdx.x;
  if (t >= NV * D) return;
  int n = t / D;
  int d = t % D;

  float R0[9], Rn[9], t0[3], tn[3];
#pragma unroll
  for (int i = 0; i < 3; ++i) {
#pragma unroll
    for (int j = 0; j < 3; ++j) {
      R0[i * 3 + j] = E[0 * 16 + i * 4 + j];
      Rn[i * 3 + j] = E[n * 16 + i * 4 + j];
    }
    t0[i] = E[0 * 16 + i * 4 + 3];
    tn[i] = E[n * 16 + i * 4 + 3];
  }
  float c0[3], cn[3], crel[3];
#pragma unroll
  for (int i = 0; i < 3; ++i) {
    c0[i] = -(R0[0 * 3 + i] * t0[0] + R0[1 * 3 + i] * t0[1] + R0[2 * 3 + i] * t0[2]);
    cn[i] = -(Rn[0 * 3 + i] * tn[0] + Rn[1 * 3 + i] * tn[1] + Rn[2 * 3 + i] * tn[2]);
    crel[i] = cn[i] - c0[i];
  }
  float fr[3] = {R0[6], R0[7], R0[8]};
  float dv = ds[0] + di[0] * (float)d;

  float mid[9];
#pragma unroll
  for (int i = 0; i < 3; ++i)
#pragma unroll
    for (int j = 0; j < 3; ++j)
      mid[i * 3 + j] = ((i == j) ? 1.0f : 0.0f) - crel[i] * fr[j] / dv;

  float A[9], K0i[9], B[9], T[9], Hm[9];
  mat3mul(Km + (size_t)n * 9, Rn, A);
  mat3inv(Km, K0i);
#pragma unroll
  for (int i = 0; i < 3; ++i)
#pragma unroll
    for (int j = 0; j < 3; ++j)
      B[i * 3 + j] = R0[0 * 3 + i] * K0i[0 * 3 + j] + R0[1 * 3 + i] * K0i[1 * 3 + j] +
                     R0[2 * 3 + i] * K0i[2 * 3 + j];
  mat3mul(A, mid, T);
  mat3mul(T, B, Hm);
  float* dst = Hs + ((size_t)n * D + d) * 9;
#pragma unroll
  for (int i = 0; i < 9; ++i) dst[i] = Hm[i];
}

// ---------------------------------------------------------------------------
// Warp src views at depth d + per-channel variance cost over the 3 views.
// Writes cost in bf16 into two destinations (the x-slot of xh1 and of cc1).
//   f  : (3, C, Hf, Wf) f32, Hs: (3, D, 3, 3)
// ---------------------------------------------------------------------------
__global__ void warp_cost_kernel(const float* __restrict__ f, const float* __restrict__ Hs,
                                 __bf16* __restrict__ o0, __bf16* __restrict__ o1,
                                 int d, int C, int Hf, int Wf, int D) {
  int idx = blockIdx.x * blockDim.x + threadIdx.x;
  int HW = Hf * Wf;
  if (idx >= HW) return;
  int y = idx / Wf;
  int x = idx - y * Wf;

  float wx[2], wy[2], vd[2];
  int base01[2];
#pragma unroll
  for (int v = 0; v < 2; ++v) {
    const float* Hm = Hs + ((size_t)(v + 1) * D + d) * 9;
    float px = Hm[0] * x + Hm[1] * y + Hm[2];
    float py = Hm[3] * x + Hm[4] * y + Hm[5];
    float pz = Hm[6] * x + Hm[7] * y + Hm[8];
    float fx = px / (pz + 1e-10f);
    float fy = py / (pz + 1e-10f);
    vd[v] = (fx >= 0.f && fx <= (float)(Wf - 1) && fy >= 0.f && fy <= (float)(Hf - 1)) ? 1.f : 0.f;
    float flx = fminf(fmaxf(floorf(fx), 0.f), (float)(Wf - 2));
    float fly = fminf(fmaxf(floorf(fy), 0.f), (float)(Hf - 2));
    int x0 = (int)flx, y0 = (int)fly;
    wx[v] = fminf(fmaxf(fx - flx, 0.f), 1.f);
    wy[v] = fminf(fmaxf(fy - fly, 0.f), 1.f);
    base01[v] = y0 * Wf + x0;
  }

  for (int c = 0; c < C; ++c) {
    float v0 = f[(size_t)c * HW + idx];  // reference view 0
    float s = v0, s2 = v0 * v0;
#pragma unroll
    for (int v = 0; v < 2; ++v) {
      const float* fc = f + ((size_t)(v + 1) * C + c) * HW;
      int b0 = base01[v];
      float f00 = fc[b0], f01 = fc[b0 + 1], f10 = fc[b0 + Wf], f11 = fc[b0 + Wf + 1];
      float o = (f00 * (1.f - wx[v]) * (1.f - wy[v]) + f01 * wx[v] * (1.f - wy[v]) +
                 f10 * (1.f - wx[v]) * wy[v] + f11 * wx[v] * wy[v]) * vd[v];
      s += o; s2 += o * o;
    }
    float m = s * (1.0f / 3.0f);
    float cst = -(s2 * (1.0f / 3.0f) - m * m);
    o0[(size_t)c * HW + idx] = (__bf16)cst;
    o1[(size_t)c * HW + idx] = (__bf16)cst;
  }
}

// ---------------------------------------------------------------------------
// GRU pointwise kernels.
// ---------------------------------------------------------------------------
__global__ void gru_rh_kernel(const float* __restrict__ rpre, const float* __restrict__ h,
                              __bf16* __restrict__ out, size_t n) {
  size_t i = (size_t)blockIdx.x * blockDim.x + threadIdx.x;
  if (i >= n) return;
  float r = 1.0f / (1.0f + expf(-rpre[i]));
  out[i] = (__bf16)(r * h[i]);
}

__global__ void gru_update_kernel(const float* __restrict__ upre, const float* __restrict__ cpre,
                                  float* __restrict__ h, size_t n) {
  size_t i = (size_t)blockIdx.x * blockDim.x + threadIdx.x;
  if (i >= n) return;
  float u = 1.0f / (1.0f + expf(-upre[i]));
  float c = tanhf(cpre[i]);
  h[i] = u * h[i] + (1.0f - u) * c;
}

// ---------------------------------------------------------------------------
// Softmax over depth + argmax -> depth image, prob image.
// ---------------------------------------------------------------------------
__global__ void depth_out_kernel(const float* __restrict__ costs, const float* __restrict__ ds,
                                 const float* __restrict__ di, float* __restrict__ out,
                                 int D, int HW) {
  int i = blockIdx.x * blockDim.x + threadIdx.x;
  if (i >= HW) return;
  float mx = -3.4e38f;
  int mi = 0;
  for (int d = 0; d < D; ++d) {
    float v = costs[(size_t)d * HW + i];
    if (v > mx) { mx = v; mi = d; }
  }
  float se = 0.f;
  for (int d = 0; d < D; ++d) se += expf(costs[(size_t)d * HW + i] - mx);
  out[i]      = ds[0] + di[0] * (float)mi;   // depth image
  out[HW + i] = 1.0f / se;                   // max softmax prob
}

// ---------------------------------------------------------------------------
// Host orchestration.
// ---------------------------------------------------------------------------
extern "C" void kernel_launch(void* const* d_in, const int* in_sizes, int n_in,
                              void* d_out, int out_size, void* d_ws, size_t ws_size,
                              hipStream_t stream) {
  (void)in_sizes; (void)n_in; (void)out_size; (void)ws_size;

  const float* images = (const float*)d_in[0];
  const float* intr   = (const float*)d_in[1];
  const float* extr   = (const float*)d_in[2];
  const float* dstart = (const float*)d_in[3];
  const float* dint   = (const float*)d_in[4];
  const int D = 64;  // depth_num (reference: fixed 64)

  // Bump allocator over the workspace (reset each call -> deterministic).
  size_t off = 0;
  auto alloc  = [&](size_t bytes) -> char* {
    char* p = (char*)d_ws + off;
    off = (off + bytes + 255) & ~(size_t)255;
    return p;
  };
  auto allocF = [&](size_t n) -> float*  { return (float*)alloc(n * sizeof(float)); };
  auto allocB = [&](size_t n) -> __bf16* { return (__bf16*)alloc(n * sizeof(__bf16)); };

  auto cvt = [&](const float* src, __bf16* dst, size_t n) {
    cvt_f32_bf16<<<cdiv_host((long long)n, 256), 256, 0, stream>>>(src, dst, n);
  };

  // ----- layer spec mirrored from the reference (dict insertion order) -----
  struct LS { char kind; int cin, cout, k, s; };
  static const LS SPEC[32] = {
      {'c', 3, 16, 3, 2},  {'c', 16, 32, 3, 2},  {'c', 32, 64, 3, 2},  {'c', 64, 128, 3, 2},
      {'c', 3, 8, 3, 1},   {'c', 8, 8, 3, 1},    {'c', 16, 16, 3, 1},  {'c', 16, 16, 3, 1},
      {'c', 32, 32, 3, 1}, {'c', 32, 32, 3, 1},  {'c', 64, 64, 3, 1},  {'c', 64, 64, 3, 1},
      {'c', 128, 128, 3, 1}, {'c', 128, 128, 3, 1},
      {'d', 128, 64, 3, 2}, {'c', 128, 64, 3, 1}, {'c', 64, 64, 3, 1},
      {'d', 64, 32, 3, 2},  {'c', 64, 32, 3, 1},  {'c', 32, 32, 3, 1},
      {'d', 32, 16, 3, 2},  {'c', 32, 16, 3, 1},  {'c', 16, 16, 3, 1},
      {'d', 16, 8, 3, 2},   {'c', 16, 8, 3, 1},   {'c', 8, 8, 3, 1},
      {'c', 8, 16, 5, 2},   {'c', 16, 16, 3, 1},  {'c', 16, 16, 3, 1},
      {'c', 16, 32, 5, 2},  {'c', 32, 32, 3, 1},  {'p', 32, 32, 3, 1}};
  int widx[32];
  {
    int w = 6;
    for (int i = 0; i < 32; ++i) { widx[i] = w; w += (SPEC[i].kind == 'p') ? 2 : 3; }
  }
  // GRU param indices: gru{1,2,3} x {r,u,c} x {w,b} starting at 101; prob at 119/120.
  int gwidx[3][3], gbidx[3][3];
  {
    int w = 101;
    for (int g = 0; g < 3; ++g)
      for (int gate = 0; gate < 3; ++gate) { gwidx[g][gate] = w; gbidx[g][gate] = w + 1; w += 2; }
  }
  const int probW = 119, probB = 120;
  const int gx[3] = {32, 16, 4};  // GRU input channels
  const int gh[3] = {16, 4, 2};   // GRU hidden channels

  // ----- convert all weights to bf16 once -----
  __bf16* wb[32];
  for (int i = 0; i < 32; ++i) {
    size_t n = (size_t)SPEC[i].cout * SPEC[i].cin * SPEC[i].k * SPEC[i].k;
    wb[i] = allocB(n);
    cvt((const float*)d_in[widx[i]], wb[i], n);
  }
  __bf16* gwb[3][3];
  for (int g = 0; g < 3; ++g)
    for (int gate = 0; gate < 3; ++gate) {
      size_t n = (size_t)gh[g] * (gx[g] + gh[g]) * 9;
      gwb[g][gate] = allocB(n);
      cvt((const float*)d_in[gwidx[g][gate]], gwb[g][gate], n);
    }
  __bf16* pwb = allocB(18);
  cvt((const float*)d_in[probW], pwb, 18);

  // ----- shared scratch -----
  float* raw  = allocF((size_t)3 * 8 * 384 * 512);  // max conv-raw tensor
  float* mu   = allocF(64);
  float* rstd = allocF(64);

  auto conv = [&](const __bf16* in, const __bf16* w, const float* bias, float* out,
                  int N, int Cin, int H, int W, int Cout, int K, int s, int pad,
                  int mode, int Ho, int Wo) {
    dim3 grid(Ho * Wo / 64, (Cout + 15) / 16, N);
    if (mode == 1) {
      conv2d_wmma<3, 1><<<grid, 128, 0, stream>>>(in, w, bias, out, Cin, H, W, Cout, s,
                                                  pad, Ho, Wo);
    } else if (K == 5) {
      conv2d_wmma<5, 0><<<grid, 128, 0, stream>>>(in, w, bias, out, Cin, H, W, Cout, s,
                                                  pad, Ho, Wo);
    } else {
      conv2d_wmma<3, 0><<<grid, 128, 0, stream>>>(in, w, bias, out, Cin, H, W, Cout, s,
                                                  pad, Ho, Wo);
    }
  };

  // conv + groupnorm + relu, emitting bf16 activations
  auto cgr = [&](int li, const __bf16* in, int N, int H, int W, int Ho, int Wo) -> __bf16* {
    const LS& L = SPEC[li];
    int mode = (L.kind == 'd') ? 1 : 0;
    int pad  = mode ? 1 : ((L.k == 5) ? 1 : ((L.s == 2) ? 0 : 1));
    conv(in, wb[li], nullptr, raw, N, L.cin, H, W, L.cout, L.k, L.s, pad, mode, Ho, Wo);
    int HW = Ho * Wo;
    int G  = L.cout / 8;
    gn_stats_kernel<<<N * G, 256, 0, stream>>>(raw, mu, rstd, L.cout, HW, G);
    size_t tot = (size_t)N * L.cout * HW;
    __bf16* outb = allocB(tot);
    gn_apply_relu<<<cdiv_host((long long)tot, 256), 256, 0, stream>>>(
        raw, (const float*)d_in[widx[li] + 1], (const float*)d_in[widx[li] + 2], mu, rstd,
        outb, L.cout, HW, G, tot);
    return outb;
  };

  auto concat = [&](const __bf16* a, const __bf16* b, int N, int C1, int C2, int HW) -> __bf16* {
    size_t tot = (size_t)N * (C1 + C2) * HW;
    __bf16* o = allocB(tot);
    concat2_bf16<<<cdiv_host((long long)tot, 256), 256, 0, stream>>>(a, b, o, C1, C2, HW, tot);
    return o;
  };

  // ----- UNet -----
  const int H0 = 384, W0 = 512, H1 = 192, W1 = 256, H2 = 96, W2 = 128, H3 = 48, W3 = 64,
            H4 = 24, W4 = 32;
  __bf16* xb = allocB((size_t)3 * 3 * H0 * W0);
  cvt(images, xb, (size_t)3 * 3 * H0 * W0);

  __bf16* c1_0 = cgr(0, xb, 3, H0, W0, H1, W1);
  __bf16* c2_0 = cgr(1, c1_0, 3, H1, W1, H2, W2);
  __bf16* c3_0 = cgr(2, c2_0, 3, H2, W2, H3, W3);
  __bf16* c4_0 = cgr(3, c3_0, 3, H3, W3, H4, W4);
  __bf16* c0_1 = cgr(4, xb, 3, H0, W0, H0, W0);
  __bf16* c0_2 = cgr(5, c0_1, 3, H0, W0, H0, W0);
  __bf16* c1_1 = cgr(6, c1_0, 3, H1, W1, H1, W1);
  __bf16* c1_2 = cgr(7, c1_1, 3, H1, W1, H1, W1);
  __bf16* c2_1 = cgr(8, c2_0, 3, H2, W2, H2, W2);
  __bf16* c2_2 = cgr(9, c2_1, 3, H2, W2, H2, W2);
  __bf16* c3_1 = cgr(10, c3_0, 3, H3, W3, H3, W3);
  __bf16* c3_2 = cgr(11, c3_1, 3, H3, W3, H3, W3);
  __bf16* c4_1 = cgr(12, c4_0, 3, H4, W4, H4, W4);
  __bf16* c4_2 = cgr(13, c4_1, 3, H4, W4, H4, W4);

  __bf16* d5   = cgr(14, c4_2, 3, H4, W4, H3, W3);
  __bf16* cat5 = concat(d5, c3_2, 3, 64, 64, H3 * W3);
  __bf16* c5_1 = cgr(15, cat5, 3, H3, W3, H3, W3);
  __bf16* c5_2 = cgr(16, c5_1, 3, H3, W3, H3, W3);

  __bf16* d6   = cgr(17, c5_2, 3, H3, W3, H2, W2);
  __bf16* cat6 = concat(d6, c2_2, 3, 32, 32, H2 * W2);
  __bf16* c6_1 = cgr(18, cat6, 3, H2, W2, H2, W2);
  __bf16* c6_2 = cgr(19, c6_1, 3, H2, W2, H2, W2);

  __bf16* d7   = cgr(20, c6_2, 3, H2, W2, H1, W1);
  __bf16* cat7 = concat(d7, c1_2, 3, 16, 16, H1 * W1);
  __bf16* c7_1 = cgr(21, cat7, 3, H1, W1, H1, W1);
  __bf16* c7_2 = cgr(22, c7_1, 3, H1, W1, H1, W1);

  __bf16* d8   = cgr(23, c7_2, 3, H1, W1, H0, W0);
  __bf16* cat8 = concat(d8, c0_2, 3, 8, 8, H0 * W0);
  __bf16* c8_1 = cgr(24, cat8, 3, H0, W0, H0, W0);
  __bf16* c8_2 = cgr(25, c8_1, 3, H0, W0, H0, W0);

  __bf16* c9_0  = cgr(26, c8_2, 3, H0, W0, H1, W1);
  __bf16* c9_1  = cgr(27, c9_0, 3, H1, W1, H1, W1);
  __bf16* c9_2  = cgr(28, c9_1, 3, H1, W1, H1, W1);
  __bf16* c10_0 = cgr(29, c9_2, 3, H1, W1, H2, W2);
  __bf16* c10_1 = cgr(30, c10_0, 3, H2, W2, H2, W2);

  // final features f = conv10_2 (bias, no GN/ReLU): (3, 32, 96, 128) f32
  float* f = allocF((size_t)3 * 32 * H2 * W2);
  conv(c10_1, wb[31], (const float*)d_in[widx[31] + 1], f, 3, 32, H2, W2, 32, 3, 1, 1, 0,
       H2, W2);

  // ----- homographies -----
  float* Hs = allocF((size_t)3 * D * 9);
  homog_kernel<<<1, 192, 0, stream>>>(intr, extr, dstart, dint, Hs, 3, D);

  // ----- GRU scan over depth -----
  const int HW = H2 * W2;  // 12288
  // xh* = [x | h] bf16 (contiguous channel concat for N=1); cc* = [x | r*h].
  __bf16* xh1 = allocB((size_t)48 * HW);
  __bf16* cc1 = allocB((size_t)48 * HW);
  __bf16* xh2 = allocB((size_t)20 * HW);
  __bf16* cc2 = allocB((size_t)20 * HW);
  __bf16* xh3 = allocB((size_t)6 * HW);
  __bf16* cc3 = allocB((size_t)6 * HW);
  float* h1 = allocF((size_t)16 * HW);
  float* h2 = allocF((size_t)4 * HW);
  float* h3 = allocF((size_t)2 * HW);
  float* rpre = allocF((size_t)16 * HW);
  float* upre = allocF((size_t)16 * HW);
  float* cpre = allocF((size_t)16 * HW);
  float* costs = allocF((size_t)D * HW);

  auto zf = [&](float* p, size_t n) { zero_f32<<<cdiv_host((long long)n, 256), 256, 0, stream>>>(p, n); };
  auto zb = [&](__bf16* p, size_t n) { zero_bf16<<<cdiv_host((long long)n, 256), 256, 0, stream>>>(p, n); };
  zf(h1, (size_t)16 * HW); zf(h2, (size_t)4 * HW); zf(h3, (size_t)2 * HW);
  zb(xh1 + (size_t)32 * HW, (size_t)16 * HW);  // h1 slot
  zb(xh2 + (size_t)16 * HW, (size_t)4 * HW);   // h2 slot
  zb(xh3 + (size_t)4 * HW, (size_t)2 * HW);    // h3 slot

  auto ew = [&](size_t n) { return dim3(cdiv_host((long long)n, 256)); };

  for (int d = 0; d < D; ++d) {
    // warp + variance cost -> x slots of xh1 and cc1 (32 channels, bf16)
    warp_cost_kernel<<<HW / 256, 256, 0, stream>>>(f, Hs, xh1, cc1, d, 32, H2, W2, D);

    // --- gru1: x=cost(32), h=16 ---
    conv(xh1, gwb[0][0], (const float*)d_in[gbidx[0][0]], rpre, 1, 48, H2, W2, 16, 3, 1, 1, 0, H2, W2);
    conv(xh1, gwb[0][1], (const float*)d_in[gbidx[0][1]], upre, 1, 48, H2, W2, 16, 3, 1, 1, 0, H2, W2);
    gru_rh_kernel<<<ew((size_t)16 * HW), 256, 0, stream>>>(rpre, h1, cc1 + (size_t)32 * HW, (size_t)16 * HW);
    conv(cc1, gwb[0][2], (const float*)d_in[gbidx[0][2]], cpre, 1, 48, H2, W2, 16, 3, 1, 1, 0, H2, W2);
    gru_update_kernel<<<ew((size_t)16 * HW), 256, 0, stream>>>(upre, cpre, h1, (size_t)16 * HW);
    cvt(h1, xh1 + (size_t)32 * HW, (size_t)16 * HW);  // next-step gru1 h slot
    cvt(h1, xh2, (size_t)16 * HW);                    // gru2 x slot
    cvt(h1, cc2, (size_t)16 * HW);                    // gru2 c-path x slot

    // --- gru2: x=h1(16), h=4 ---
    conv(xh2, gwb[1][0], (const float*)d_in[gbidx[1][0]], rpre, 1, 20, H2, W2, 4, 3, 1, 1, 0, H2, W2);
    conv(xh2, gwb[1][1], (const float*)d_in[gbidx[1][1]], upre, 1, 20, H2, W2, 4, 3, 1, 1, 0, H2, W2);
    gru_rh_kernel<<<ew((size_t)4 * HW), 256, 0, stream>>>(rpre, h2, cc2 + (size_t)16 * HW, (size_t)4 * HW);
    conv(cc2, gwb[1][2], (const float*)d_in[gbidx[1][2]], cpre, 1, 20, H2, W2, 4, 3, 1, 1, 0, H2, W2);
    gru_update_kernel<<<ew((size_t)4 * HW), 256, 0, stream>>>(upre, cpre, h2, (size_t)4 * HW);
    cvt(h2, xh2 + (size_t)16 * HW, (size_t)4 * HW);
    cvt(h2, xh3, (size_t)4 * HW);
    cvt(h2, cc3, (size_t)4 * HW);

    // --- gru3: x=h2(4), h=2 ---
    conv(xh3, gwb[2][0], (const float*)d_in[gbidx[2][0]], rpre, 1, 6, H2, W2, 2, 3, 1, 1, 0, H2, W2);
    conv(xh3, gwb[2][1], (const float*)d_in[gbidx[2][1]], upre, 1, 6, H2, W2, 2, 3, 1, 1, 0, H2, W2);
    gru_rh_kernel<<<ew((size_t)2 * HW), 256, 0, stream>>>(rpre, h3, cc3 + (size_t)4 * HW, (size_t)2 * HW);
    conv(cc3, gwb[2][2], (const float*)d_in[gbidx[2][2]], cpre, 1, 6, H2, W2, 2, 3, 1, 1, 0, H2, W2);
    gru_update_kernel<<<ew((size_t)2 * HW), 256, 0, stream>>>(upre, cpre, h3, (size_t)2 * HW);
    cvt(h3, xh3 + (size_t)4 * HW, (size_t)2 * HW);  // also the prob-conv input

    // --- prob conv (2 -> 1 channels) -> costs[d] ---
    conv(xh3 + (size_t)4 * HW, pwb, (const float*)d_in[probB], costs + (size_t)d * HW,
         1, 2, H2, W2, 1, 3, 1, 1, 0, H2, W2);
  }

  // ----- softmax over depth + argmax -> (depth_image, prob_image) -----
  depth_out_kernel<<<cdiv_host(HW, 256), 256, 0, stream>>>(costs, dstart, dint,
                                                           (float*)d_out, D, HW);
}
